// EvoformerEncoderLayer_85813446574724
// MI455X (gfx1250) — matrix operand order repaired and checked by reading, not compiled, not run on hardware
//
#include <hip/hip_runtime.h>
#include <cstdint>
#include <math.h>

typedef __attribute__((ext_vector_type(16))) __bf16 v16bf;
typedef __attribute__((ext_vector_type(8)))  float  v8f;

static constexpr int  BB    = 2;
static constexpr int  Nn    = 384;
static constexpr int  DNODE = 256;
static constexpr int  DEDGE = 128;
static constexpr long NN2   = (long)Nn * Nn;     // 147456
static constexpr long NEDGE = (long)BB * NN2;    // 294912

// ---------------------------------------------------------------------------
// WMMA fragment helpers (CDNA5 16x16x32 bf16 layouts, wave32)
// A 16x32: lane m (&15) holds row m; khalf=(lane>>4): elems 0..7 -> K=k0+8*kh+0..7,
//          elems 8..15 -> K=k0+8*kh+16..23  => two contiguous 16B chunks.
// B 32x16 taken from Wt[n][k] (row-major transposed weights): lane holds col n=lane&15,
//          same K order as A: elems 0..7 -> K=k0+8*kh+0..7, elems 8..15 -> K=k0+8*kh+16..23.
// C/D 16x16 f32: vgpr v -> row = v + 8*(lane>>4), col = lane&15.
// ---------------------------------------------------------------------------
union BF16x16 { uint4 u[2]; v16bf v; };

__device__ __forceinline__ v16bf load_a_frag(const __bf16* A, int lda, long m0, int k0, int lane) {
    const __bf16* p = A + (m0 + (lane & 15)) * (long)lda + k0 + ((lane >> 4) << 3);
    BF16x16 t;
    t.u[0] = *(const uint4*)(p);
    t.u[1] = *(const uint4*)(p + 16);
    return t.v;
}
__device__ __forceinline__ v16bf load_b_frag(const __bf16* W, int ldw, int n0, int k0, int lane) {
    const __bf16* p = W + (long)(n0 + (lane & 15)) * ldw + k0 + ((lane >> 4) << 3);
    BF16x16 t;
    t.u[0] = *(const uint4*)(p);
    t.u[1] = *(const uint4*)(p + 16);
    return t.v;
}
__device__ __forceinline__ v8f wmma_bf16(v16bf a, v16bf b, v8f c) {
    return __builtin_amdgcn_wmma_f32_16x16x32_bf16(false, a, false, b, (short)0, c, false, false);
}
__device__ __forceinline__ float sigmoidf_(float x) { return 1.f / (1.f + expf(-x)); }

// ---------------------------------------------------------------------------
// Generic tiled GEMM: out = A[M,K] @ Wt[N,K]^T (+ C) with per-EPI epilogue.
// Block = 256 thr = 8 waves (4 M x 2 N). Non-dual: wave does 16x32 (tiles n0, n0+16).
// Dual (EPI 8): wave does value tile n0 and gate tile n0+64 (for sigmoid gating).
// Tile guards are read-first-laned so they are scalar branches (EXEC stays all-1s
// around every WMMA, per ISA requirement) and are hoisted out of the K loop.
// ---------------------------------------------------------------------------
struct GemmP {
    const __bf16* A; const __bf16* W;
    int lda, ldw, K, N;
    long zA, zW; int zdivA, zdivW;
    const float* C; long zC; int ldc;
    const float* bias; const float* bias2;
    float* outf; long zO; int ldo;
    __bf16* outb; __bf16* outb2; int ldob;
    float* ecur; const __bf16* gbuf;
    const float* resid; const uint8_t* mask;
    float scale;
};

template<int EPI>
__device__ __forceinline__ void gemm_store(const GemmP& p, int z, long tileM, int n0, int lane, v8f acc) {
    int col = n0 + (lane & 15);
    #pragma unroll
    for (int v = 0; v < 8; ++v) {
        long row = tileM + v + ((lane >> 4) << 3);
        float x = acc[v];
        if (EPI == 0) {                     // plain f32 store (scores / triangle prod)
            p.outf[(long)z * p.zO + row * (long)p.ldo + col] = x;
        } else if (EPI == 1) {              // q/k: bf16 [B*H][N][32], scaled
            long b = row / Nn, i = row % Nn;
            int h = col >> 5, d = col & 31;
            p.outb[(((b * 8 + h) * Nn) + i) * 32 + d] = (__bf16)((x + p.bias[col]) * p.scale);
        } else if (EPI == 2) {              // v transposed: bf16 [B*H][32][N]
            long b = row / Nn, i = row % Nn;
            int h = col >> 5, d = col & 31;
            p.outb[(((b * 8 + h) * 32) + d) * Nn + i] = (__bf16)(x + p.bias[col]);
        } else if (EPI == 3) {              // attn@v heads -> bf16 [B*N][256]; z=(b,h)
            long b = z >> 3; int h = z & 7;
            p.outb[((b * Nn + row) * DNODE) + h * 32 + col] = (__bf16)x;
        } else if (EPI == 4) {              // Xcur = X + acc + bias
            long idx = row * DNODE + col;
            p.outf[idx] = p.resid[idx] + x + p.bias[col];
        } else if (EPI == 5) {              // masked bf16 (+bias)  (OPB a/b)
            float m = p.mask[row] ? 1.f : 0.f;
            p.outb[row * (long)p.ldob + col] = (__bf16)((x + p.bias[col]) * m);
        } else if (EPI == 6) {              // plain bf16 row-major (T tensor)
            float bb = p.bias ? p.bias[col] : 0.f;
            p.outb[row * (long)p.ldob + col] = (__bf16)(x + bb);
        } else if (EPI == 7) {              // OPB edge init: Ecur = E + acc + bias; z=(b,i), row=j
            long idx = ((long)z * Nn + row) * DEDGE + col;
            p.ecur[idx] = p.resid[idx] + x + p.bias[col];
        } else if (EPI == 9) {              // sigmoid bf16 (TMB gate g)
            p.outb[row * (long)p.ldob + col] = (__bf16)sigmoidf_(x + p.bias[col]);
        } else if (EPI == 10) {             // Ecur += g * (acc + bias)
            long idx = row * (long)DEDGE + col;
            p.ecur[idx] += (float)p.gbuf[idx] * (x + p.bias[col]);
        }
    }
}

template<int EPI, bool LOADC, bool DUAL>
__global__ void gemm_k(GemmP p) {
    int lane = threadIdx.x & 31;
    // wave-uniform tile coordinates, forced into SGPRs so guards are scalar branches
    int wave = __builtin_amdgcn_readfirstlane(threadIdx.x >> 5);
    int wm = wave & 3, wn = wave >> 2;
    int z = blockIdx.z;
    const __bf16* A = p.A + (long)(z / p.zdivA) * p.zA;
    const __bf16* W = p.W + (long)(z / p.zdivW) * p.zW;
    long tileM = (long)blockIdx.y * 64 + wm * 16;
    constexpr int NOFF = DUAL ? 64 : 16;
    int n0 = DUAL ? ((int)blockIdx.x * 32 + wn * 16) : ((int)blockIdx.x * 64 + wn * 32);
    bool do0 = n0 < p.N;                          // scalar
    bool do1 = DUAL || (n0 + 16) < p.N;           // scalar; non-dual: do1 => do0
    if (!do0) return;

    v8f acc0 = {}; v8f acc1 = {};
    if (LOADC) {
        const float* C = p.C + (long)z * p.zC;
        #pragma unroll
        for (int v = 0; v < 8; ++v) {
            long row = tileM + v + ((lane >> 4) << 3);
            acc0[v] = C[row * (long)p.ldc + n0 + (lane & 15)];
            if (do1) acc1[v] = C[row * (long)p.ldc + n0 + NOFF + (lane & 15)];
        }
    }
    if (do1) {   // dual-tile hot loop: pure loads + 2x WMMA, no EXEC churn
        for (int k = 0; k < p.K; k += 32) {
            v16bf a = load_a_frag(A, p.lda, tileM, k, lane);
            acc0 = wmma_bf16(a, load_b_frag(W, p.ldw, n0,        k, lane), acc0);
            acc1 = wmma_bf16(a, load_b_frag(W, p.ldw, n0 + NOFF, k, lane), acc1);
        }
    } else {     // single-tile loop
        for (int k = 0; k < p.K; k += 32) {
            v16bf a = load_a_frag(A, p.lda, tileM, k, lane);
            acc0 = wmma_bf16(a, load_b_frag(W, p.ldw, n0, k, lane), acc0);
        }
    }

    if (DUAL) {  // EPI 8: ta/tb = mask * sigmoid(gate) * value; scatter to row+transposed layouts
        int col = n0 + (lane & 15);               // channel c in [0,64)
        #pragma unroll
        for (int v = 0; v < 8; ++v) {
            long r = tileM + v + ((lane >> 4) << 3);    // edge-flat row (b,i,k)
            long b = r / NN2; long rem = r % NN2;
            long i = rem / Nn, kk = rem % Nn;
            float val = acc0[v] + p.bias[col];
            float gt  = acc1[v] + p.bias2[col];
            float m   = p.mask[r] ? 1.f : 0.f;
            float t   = m * val * sigmoidf_(gt);
            p.outb [(((b * 64 + col) * Nn) + i)  * Nn + kk] = (__bf16)t;  // [b][c][i][k]
            p.outb2[(((b * 64 + col) * Nn) + kk) * Nn + i]  = (__bf16)t;  // [b][c][k][i]
        }
    } else {
        gemm_store<EPI>(p, z, tileM, n0, lane, acc0);
        if (do1) gemm_store<EPI>(p, z, tileM, n0 + 16, lane, acc1);
    }
}

// ---------------------------------------------------------------------------
// Fused FFN: 16 rows per block. LN -> LDS(bf16) -> GEMM1+ReLU -> LDS -> GEMM2 + residual.
// ---------------------------------------------------------------------------
template<int DIN, int DHID>
__global__ void fused_ffn_k(const float* __restrict__ src,
                            const float* __restrict__ ln_s, const float* __restrict__ ln_b,
                            const __bf16* __restrict__ W1t, const float* __restrict__ b1,
                            const __bf16* __restrict__ W2t, const float* __restrict__ b2,
                            float* __restrict__ dst) {
    __shared__ __bf16 lnrow[16 * DIN];
    __shared__ __bf16 hid[16 * DHID];
    long r0 = (long)blockIdx.x * 16;
    int lane = threadIdx.x & 31;
    int wave = __builtin_amdgcn_readfirstlane(threadIdx.x >> 5);
    for (int rr = wave * 2; rr < wave * 2 + 2; ++rr) {
        const float* x = src + (r0 + rr) * DIN;
        float s = 0.f, sq = 0.f;
        for (int i = lane; i < DIN; i += 32) { float v = x[i]; s += v; sq += v * v; }
        for (int o = 16; o; o >>= 1) { s += __shfl_xor(s, o, 32); sq += __shfl_xor(sq, o, 32); }
        float m = s / DIN; float var = sq / DIN - m * m;
        float rs = rsqrtf(var + 1e-5f);
        for (int i = lane; i < DIN; i += 32)
            lnrow[rr * DIN + i] = (__bf16)((x[i] - m) * rs * ln_s[i] + ln_b[i]);
    }
    __syncthreads();
    for (int t = wave; t < DHID / 16; t += 8) {
        int n0 = t * 16;
        v8f acc = {};
        #pragma unroll
        for (int k = 0; k < DIN; k += 32)
            acc = wmma_bf16(load_a_frag((const __bf16*)lnrow, DIN, 0, k, lane),
                            load_b_frag(W1t, DIN, n0, k, lane), acc);
        int col = n0 + (lane & 15);
        #pragma unroll
        for (int v = 0; v < 8; ++v) {
            int row = v + ((lane >> 4) << 3);
            float x = acc[v] + b1[col];
            hid[row * DHID + col] = (__bf16)(x > 0.f ? x : 0.f);
        }
    }
    __syncthreads();
    for (int t = wave; t < DIN / 16; t += 8) {
        int n0 = t * 16;
        v8f acc = {};
        for (int k = 0; k < DHID; k += 32)
            acc = wmma_bf16(load_a_frag((const __bf16*)hid, DHID, 0, k, lane),
                            load_b_frag(W2t, DHID, n0, k, lane), acc);
        int col = n0 + (lane & 15);
        #pragma unroll
        for (int v = 0; v < 8; ++v) {
            int row = v + ((lane >> 4) << 3);
            long idx = (r0 + row) * DIN + col;
            dst[idx] = src[idx] + acc[v] + b2[col];
        }
    }
}

// ---------------------------------------------------------------------------
// Row LayerNorm -> bf16 (one wave per row)
// ---------------------------------------------------------------------------
__global__ void rowln_k(const float* __restrict__ src, const float* __restrict__ s,
                        const float* __restrict__ b, __bf16* __restrict__ dst,
                        int C, long rows) {
    long r = (long)blockIdx.x * 8 + (threadIdx.x >> 5);
    if (r >= rows) return;
    int lane = threadIdx.x & 31;
    const float* x = src + r * C;
    float sum = 0.f, sq = 0.f;
    for (int i = lane; i < C; i += 32) { float v = x[i]; sum += v; sq += v * v; }
    for (int o = 16; o; o >>= 1) { sum += __shfl_xor(sum, o, 32); sq += __shfl_xor(sq, o, 32); }
    float m = sum / C, var = sq / C - m * m;
    float rs = rsqrtf(var + 1e-5f);
    __bf16* d = dst + r * C;
    for (int i = lane; i < C; i += 32) d[i] = (__bf16)((x[i] - m) * rs * s[i] + b[i]);
}

// LN over 64 channels of prod stored channel-major [b][c][i][j] -> lnprod bf16 [r][64]
__global__ void lnprod_k(const float* __restrict__ prod, const float* __restrict__ s,
                         const float* __restrict__ b, __bf16* __restrict__ dst) {
    long r = (long)blockIdx.x * 8 + (threadIdx.x >> 5);
    if (r >= NEDGE) return;
    int lane = threadIdx.x & 31;
    long bb = r / NN2, ij = r % NN2;
    const float* base = prod + bb * 64 * NN2 + ij;
    float v0 = base[(long)lane * NN2];
    float v1 = base[(long)(lane + 32) * NN2];
    float sum = v0 + v1, sq = v0 * v0 + v1 * v1;
    for (int o = 16; o; o >>= 1) { sum += __shfl_xor(sum, o, 32); sq += __shfl_xor(sq, o, 32); }
    float m = sum / 64.f, var = sq / 64.f - m * m;
    float rs = rsqrtf(var + 1e-5f);
    __bf16* d = dst + r * 64;
    d[lane]      = (__bf16)((v0 - m) * rs * s[lane]      + b[lane]);
    d[lane + 32] = (__bf16)((v1 - m) * rs * s[lane + 32] + b[lane + 32]);
}

// Pair bias + key-mask into the score buffer (WMMA C-init of QK^T)
__global__ void pairbias_k(const float* __restrict__ E, const float* __restrict__ pbW,
                           const float* __restrict__ pbb, const uint8_t* __restrict__ mN,
                           float* __restrict__ scores) {
    long r = (long)blockIdx.x * 8 + (threadIdx.x >> 5);
    if (r >= NEDGE) return;
    int lane = threadIdx.x & 31;
    const float* e = E + r * DEDGE;
    float acc[8] = {0.f, 0.f, 0.f, 0.f, 0.f, 0.f, 0.f, 0.f};
    #pragma unroll
    for (int kk = 0; kk < 4; ++kk) {
        int k = lane * 4 + kk;
        float ev = e[k];
        #pragma unroll
        for (int h = 0; h < 8; ++h) acc[h] += ev * pbW[k * 8 + h];
    }
    #pragma unroll
    for (int h = 0; h < 8; ++h)
        for (int o = 16; o; o >>= 1) acc[h] += __shfl_xor(acc[h], o, 32);
    long b = r / NN2; long rem = r % NN2;
    long i = rem / Nn, j = rem % Nn;
    if (lane < 8) {
        float v = mN[b * Nn + j] ? (acc[lane] + pbb[lane]) : -1e9f;
        scores[(((b * 8 + lane) * Nn) + i) * Nn + j] = v;
    }
}

__global__ void softmax_k(const float* __restrict__ sc, __bf16* __restrict__ attn) {
    long r = (long)blockIdx.x * 8 + (threadIdx.x >> 5);
    if (r >= (long)BB * 8 * Nn) return;
    int lane = threadIdx.x & 31;
    const float* s = sc + r * Nn;
    float v[12]; float mx = -3.0e38f;
    #pragma unroll
    for (int i = 0; i < 12; ++i) { v[i] = s[lane + 32 * i]; mx = fmaxf(mx, v[i]); }
    for (int o = 16; o; o >>= 1) mx = fmaxf(mx, __shfl_xor(mx, o, 32));
    float sum = 0.f;
    #pragma unroll
    for (int i = 0; i < 12; ++i) { v[i] = expf(v[i] - mx); sum += v[i]; }
    for (int o = 16; o; o >>= 1) sum += __shfl_xor(sum, o, 32);
    float inv = 1.f / sum;
    __bf16* d = attn + r * Nn;
    #pragma unroll
    for (int i = 0; i < 12; ++i) d[lane + 32 * i] = (__bf16)(v[i] * inv);
}

// Weight packers: Wt[n][k] = W[k][n], f32 -> bf16
__global__ void packT_k(const float* __restrict__ W, __bf16* __restrict__ Wt, int K, int Nout) {
    long total = (long)K * Nout;
    for (long idx = (long)blockIdx.x * 256 + threadIdx.x; idx < total; idx += (long)gridDim.x * 256) {
        long n = idx / K, k = idx % K;
        Wt[idx] = (__bf16)W[k * Nout + n];
    }
}
// concat (value||gate): Wt[128][128], rows 0..63 from Wv(128x64), 64..127 from Wg
__global__ void packcat_k(const float* __restrict__ Wv, const float* __restrict__ Wg,
                          __bf16* __restrict__ Wt) {
    int idx = blockIdx.x * 256 + threadIdx.x;
    if (idx >= 128 * 128) return;
    int n = idx / 128, k = idx % 128;
    Wt[idx] = (__bf16)((n < 64) ? Wv[k * 64 + n] : Wg[k * 64 + (n - 64)]);
}
// opb_Wout (32,32,128) -> Wt[(e*32+q)][p]
__global__ void pack_opbwout_k(const float* __restrict__ W, __bf16* __restrict__ Wt) {
    int idx = blockIdx.x * 256 + threadIdx.x;
    if (idx >= 4096 * 32) return;
    int pp = idx & 31; int n = idx >> 5; int q = n & 31; int e = n >> 5;
    Wt[idx] = (__bf16)W[pp * 4096 + q * 128 + e];
}

// ---------------------------------------------------------------------------
extern "C" void kernel_launch(void* const* d_in, const int* in_sizes, int n_in,
                              void* d_out, int out_size, void* d_ws, size_t ws_size,
                              hipStream_t stream) {
    (void)in_sizes; (void)n_in; (void)out_size; (void)ws_size;
    const float* X = (const float*)d_in[0];
    const float* E = (const float*)d_in[1];
    const uint8_t* mN = (const uint8_t*)d_in[2];
    const uint8_t* mE = (const uint8_t*)d_in[3];
    enum { i_pb_W = 0, i_pb_b, i_sab_ln_s, i_sab_ln_b, i_Wq, i_bq, i_Wk, i_bk, i_Wv, i_bv,
           i_Wo, i_bo, i_ffn_ln_s, i_ffn_ln_b, i_ffn_W1, i_ffn_b1, i_ffn_W2, i_ffn_b2,
           i_opb_ln_s, i_opb_ln_b, i_opb_W1, i_opb_b1, i_opb_W2, i_opb_b2, i_opb_Wout, i_opb_bout,
           i_tm_ln_s, i_tm_ln_b, i_tm_Wa, i_tm_ba, i_tm_Wag, i_tm_bag, i_tm_Wb, i_tm_bb,
           i_tm_Wbg, i_tm_bbg, i_tm_Wg, i_tm_bg, i_tm_ln2_s, i_tm_ln2_b, i_tm_Wout, i_tm_bout,
           i_ffe_ln_s, i_ffe_ln_b, i_ffe_W1, i_ffe_b1, i_ffe_W2, i_ffe_b2 };
    const float* P[48];
    for (int i = 0; i < 48; ++i) P[i] = (const float*)d_in[4 + i];

    // ---- workspace layout ----
    char* wsb = (char*)d_ws; size_t off = 0;
    auto alloc = [&](size_t bytes) -> void* {
        void* r = wsb + off; off = (off + bytes + 255) & ~(size_t)255; return r;
    };
    __bf16* wq_t   = (__bf16*)alloc(256 * 256 * 2);
    __bf16* wk_t   = (__bf16*)alloc(256 * 256 * 2);
    __bf16* wv_t   = (__bf16*)alloc(256 * 256 * 2);
    __bf16* wo_t   = (__bf16*)alloc(256 * 256 * 2);
    __bf16* wffn1  = (__bf16*)alloc(1024 * 256 * 2);
    __bf16* wffn2  = (__bf16*)alloc(256 * 1024 * 2);
    __bf16* wopb1  = (__bf16*)alloc(32 * 256 * 2);
    __bf16* wopb2  = (__bf16*)alloc(32 * 256 * 2);
    __bf16* wopbo  = (__bf16*)alloc(4096 * 32 * 2);
    __bf16* wtmA   = (__bf16*)alloc(128 * 128 * 2);
    __bf16* wtmB   = (__bf16*)alloc(128 * 128 * 2);
    __bf16* wtmG   = (__bf16*)alloc(128 * 128 * 2);
    __bf16* wtmO   = (__bf16*)alloc(128 * 64 * 2);
    __bf16* wffe1  = (__bf16*)alloc(512 * 128 * 2);
    __bf16* wffe2  = (__bf16*)alloc(128 * 512 * 2);
    __bf16* Xln    = (__bf16*)alloc((size_t)768 * 256 * 2);
    __bf16* qb     = (__bf16*)alloc((size_t)16 * 384 * 32 * 2);
    __bf16* kb     = (__bf16*)alloc((size_t)16 * 384 * 32 * 2);
    __bf16* vt     = (__bf16*)alloc((size_t)16 * 32 * 384 * 2);
    __bf16* ob     = (__bf16*)alloc((size_t)768 * 256 * 2);
    float*  scores = (float*) alloc((size_t)16 * NN2 * 4);
    __bf16* attn   = (__bf16*)alloc((size_t)16 * NN2 * 2);
    float*  Xcur   = (float*) alloc((size_t)768 * 256 * 4);
    __bf16* abuf   = (__bf16*)alloc((size_t)768 * 32 * 2);
    __bf16* bbuf   = (__bf16*)alloc((size_t)768 * 32 * 2);
    __bf16* Tt     = (__bf16*)alloc((size_t)768 * 4096 * 2);
    __bf16* hE     = (__bf16*)alloc((size_t)NEDGE * 128 * 2);   // later aliased as f32 prod
    __bf16* TAr    = (__bf16*)alloc((size_t)BB * 64 * NN2 * 2); // later aliased as lnprod
    __bf16* TAt    = (__bf16*)alloc((size_t)BB * 64 * NN2 * 2);
    __bf16* TBr    = (__bf16*)alloc((size_t)BB * 64 * NN2 * 2);
    __bf16* TBt    = (__bf16*)alloc((size_t)BB * 64 * NN2 * 2);
    __bf16* gbuf   = (__bf16*)alloc((size_t)NEDGE * 128 * 2);
    float*  prod   = (float*)hE;      // reuse after edge projections are done
    __bf16* lnprod = TAr;             // reuse after triangle GEMMs are done

    float* Xout = (float*)d_out;                  // final X  [768,256]
    float* Ecur = (float*)d_out + 768 * 256;      // E working/final buffer [NE,128]

    dim3 blk(256);

    // ---- 0) pack weights (transposed bf16) ----
    packT_k<<<dim3(256), blk, 0, stream>>>(P[i_Wq], wq_t, 256, 256);
    packT_k<<<dim3(256), blk, 0, stream>>>(P[i_Wk], wk_t, 256, 256);
    packT_k<<<dim3(256), blk, 0, stream>>>(P[i_Wv], wv_t, 256, 256);
    packT_k<<<dim3(256), blk, 0, stream>>>(P[i_Wo], wo_t, 256, 256);
    packT_k<<<dim3(1024), blk, 0, stream>>>(P[i_ffn_W1], wffn1, 256, 1024);
    packT_k<<<dim3(1024), blk, 0, stream>>>(P[i_ffn_W2], wffn2, 1024, 256);
    packT_k<<<dim3(32),  blk, 0, stream>>>(P[i_opb_W1], wopb1, 256, 32);
    packT_k<<<dim3(32),  blk, 0, stream>>>(P[i_opb_W2], wopb2, 256, 32);
    pack_opbwout_k<<<dim3(512), blk, 0, stream>>>(P[i_opb_Wout], wopbo);
    packcat_k<<<dim3(64), blk, 0, stream>>>(P[i_tm_Wa], P[i_tm_Wag], wtmA);
    packcat_k<<<dim3(64), blk, 0, stream>>>(P[i_tm_Wb], P[i_tm_Wbg], wtmB);
    packT_k<<<dim3(64),  blk, 0, stream>>>(P[i_tm_Wg], wtmG, 128, 128);
    packT_k<<<dim3(32),  blk, 0, stream>>>(P[i_tm_Wout], wtmO, 64, 128);
    packT_k<<<dim3(256), blk, 0, stream>>>(P[i_ffe_W1], wffe1, 128, 512);
    packT_k<<<dim3(256), blk, 0, stream>>>(P[i_ffe_W2], wffe2, 512, 128);

    auto zp = []() { GemmP p{}; p.zdivA = 1; p.zdivW = 1; return p; };

    // ---- 1) SAB ----
    rowln_k<<<dim3(96), blk, 0, stream>>>(X, P[i_sab_ln_s], P[i_sab_ln_b], Xln, 256, 768);
    { GemmP p = zp(); p.A = Xln; p.lda = 256; p.W = wq_t; p.ldw = 256; p.K = 256; p.N = 256;
      p.bias = P[i_bq]; p.outb = qb; p.scale = 0.1767766953f;   // 1/sqrt(32)
      gemm_k<1, false, false><<<dim3(4, 12, 1), blk, 0, stream>>>(p); }
    { GemmP p = zp(); p.A = Xln; p.lda = 256; p.W = wk_t; p.ldw = 256; p.K = 256; p.N = 256;
      p.bias = P[i_bk]; p.outb = kb; p.scale = 1.f;
      gemm_k<1, false, false><<<dim3(4, 12, 1), blk, 0, stream>>>(p); }
    { GemmP p = zp(); p.A = Xln; p.lda = 256; p.W = wv_t; p.ldw = 256; p.K = 256; p.N = 256;
      p.bias = P[i_bv]; p.outb = vt;
      gemm_k<2, false, false><<<dim3(4, 12, 1), blk, 0, stream>>>(p); }
    pairbias_k<<<dim3(36864), blk, 0, stream>>>(E, P[i_pb_W], P[i_pb_b], mN, scores);
    { GemmP p = zp(); p.A = qb; p.lda = 32; p.zA = 12288; p.W = kb; p.ldw = 32; p.zW = 12288;
      p.K = 32; p.N = 384; p.C = scores; p.zC = NN2; p.ldc = 384;
      p.outf = scores; p.zO = NN2; p.ldo = 384;
      gemm_k<0, true, false><<<dim3(6, 6, 16), blk, 0, stream>>>(p); }
    softmax_k<<<dim3(768), blk, 0, stream>>>(scores, attn);
    { GemmP p = zp(); p.A = attn; p.lda = 384; p.zA = NN2; p.W = vt; p.ldw = 384; p.zW = 12288;
      p.K = 384; p.N = 32; p.outb = ob;
      gemm_k<3, false, false><<<dim3(1, 6, 16), blk, 0, stream>>>(p); }
    { GemmP p = zp(); p.A = ob; p.lda = 256; p.W = wo_t; p.ldw = 256; p.K = 256; p.N = 256;
      p.bias = P[i_bo]; p.resid = X; p.outf = Xcur;
      gemm_k<4, false, false><<<dim3(4, 12, 1), blk, 0, stream>>>(p); }

    // ---- 2) node FFN (fused), writes final X to d_out ----
    fused_ffn_k<256, 1024><<<dim3(48), blk, 0, stream>>>(
        Xcur, P[i_ffn_ln_s], P[i_ffn_ln_b], wffn1, P[i_ffn_b1], wffn2, P[i_ffn_b2], Xout);

    // ---- 3) OPB: outer-product edge update ----
    rowln_k<<<dim3(96), blk, 0, stream>>>(Xout, P[i_opb_ln_s], P[i_opb_ln_b], Xln, 256, 768);
    { GemmP p = zp(); p.A = Xln; p.lda = 256; p.W = wopb1; p.ldw = 256; p.K = 256; p.N = 32;
      p.bias = P[i_opb_b1]; p.mask = mN; p.outb = abuf; p.ldob = 32;
      gemm_k<5, false, false><<<dim3(1, 12, 1), blk, 0, stream>>>(p); }
    { GemmP p = zp(); p.A = Xln; p.lda = 256; p.W = wopb2; p.ldw = 256; p.K = 256; p.N = 32;
      p.bias = P[i_opb_b2]; p.mask = mN; p.outb = bbuf; p.ldob = 32;
      gemm_k<5, false, false><<<dim3(1, 12, 1), blk, 0, stream>>>(p); }
    { GemmP p = zp(); p.A = abuf; p.lda = 32; p.W = wopbo; p.ldw = 32; p.K = 32; p.N = 4096;
      p.outb = Tt; p.ldob = 4096;   // T_t[b,i][e*32+q]
      gemm_k<6, false, false><<<dim3(64, 12, 1), blk, 0, stream>>>(p); }
    { GemmP p = zp(); p.A = bbuf; p.lda = 32; p.zdivA = 384; p.zA = 384 * 32;
      p.W = Tt; p.ldw = 32; p.zW = 4096; p.K = 32; p.N = 128;
      p.bias = P[i_opb_bout]; p.resid = E; p.ecur = Ecur;
      gemm_k<7, false, false><<<dim3(2, 6, 768), blk, 0, stream>>>(p); }

    // ---- 4) TMB: triangular multiplication ----
    rowln_k<<<dim3(36864), blk, 0, stream>>>(Ecur, P[i_tm_ln_s], P[i_tm_ln_b], hE, 128, NEDGE);
    { GemmP p = zp(); p.A = hE; p.lda = 128; p.W = wtmA; p.ldw = 128; p.K = 128; p.N = 64;
      p.bias = P[i_tm_ba]; p.bias2 = P[i_tm_bag]; p.mask = mE; p.outb = TAr; p.outb2 = TAt;
      gemm_k<8, false, true><<<dim3(2, 4608, 1), blk, 0, stream>>>(p); }
    { GemmP p = zp(); p.A = hE; p.lda = 128; p.W = wtmB; p.ldw = 128; p.K = 128; p.N = 64;
      p.bias = P[i_tm_bb]; p.bias2 = P[i_tm_bbg]; p.mask = mE; p.outb = TBr; p.outb2 = TBt;
      gemm_k<8, false, true><<<dim3(2, 4608, 1), blk, 0, stream>>>(p); }
    { GemmP p = zp(); p.A = hE; p.lda = 128; p.W = wtmG; p.ldw = 128; p.K = 128; p.N = 128;
      p.bias = P[i_tm_bg]; p.outb = gbuf; p.ldob = 128;
      gemm_k<9, false, false><<<dim3(2, 4608, 1), blk, 0, stream>>>(p); }
    // prod overwrites hE storage from here on
    { GemmP p = zp(); p.A = TAr; p.lda = 384; p.zA = NN2; p.W = TBr; p.ldw = 384; p.zW = NN2;
      p.K = 384; p.N = 384; p.outf = prod; p.zO = NN2; p.ldo = 384;
      gemm_k<0, false, false><<<dim3(6, 6, 128), blk, 0, stream>>>(p); }
    { GemmP p = zp(); p.A = TAt; p.lda = 384; p.zA = NN2; p.W = TBt; p.ldw = 384; p.zW = NN2;
      p.K = 384; p.N = 384; p.C = prod; p.zC = NN2; p.ldc = 384;
      p.outf = prod; p.zO = NN2; p.ldo = 384;
      gemm_k<0, true, false><<<dim3(6, 6, 128), blk, 0, stream>>>(p); }
    lnprod_k<<<dim3(36864), blk, 0, stream>>>(prod, P[i_tm_ln2_s], P[i_tm_ln2_b], lnprod);
    { GemmP p = zp(); p.A = lnprod; p.lda = 64; p.W = wtmO; p.ldw = 64; p.K = 64; p.N = 128;
      p.bias = P[i_tm_bout]; p.gbuf = gbuf; p.ecur = Ecur;
      gemm_k<10, false, false><<<dim3(2, 4608, 1), blk, 0, stream>>>(p); }

    // ---- 5) edge FFN (fused, in place in d_out) ----
    fused_ffn_k<128, 512><<<dim3(18432), blk, 0, stream>>>(
        Ecur, P[i_ffe_ln_s], P[i_ffe_ln_b], wffe1, P[i_ffe_b1], wffe2, P[i_ffe_b2], Ecur);
}
